// CirculateSwinBlock_13391708029389
// MI455X (gfx1250) — matrix operand order repaired and checked by reading, not compiled, not run on hardware
//
#include <hip/hip_runtime.h>

// ---------------------------------------------------------------------------
// CirculateSwinAttention for MI455X (gfx1250). All matmuls via
// v_wmma_f32_16x16x32_bf16 (wave32 WMMA), bf16 A/B with fp32 accumulate.
// Weights pre-converted to bf16; all fragments are b128 loads; B-fragment
// loads clustered ahead of WMMA bursts; A-fragments hoisted and reused.
// ---------------------------------------------------------------------------

#define BB    8
#define NN    8192
#define DIMC  256
#define HEADS 8
#define DKEY  64
#define WINT  128
#define NRPF  480
#define HDC   512      // HEADS*DKEY
#define NLAY  4
#define HALFW 64
#define NWIN  64       // NN / WINT
#define PREL  255      // 2*WINT-1

typedef __attribute__((ext_vector_type(16))) __bf16 v16bf;
typedef __attribute__((ext_vector_type(8)))  float  v8f;

union FragBF { v16bf v; unsigned int u[8]; unsigned short h[16]; };
union FragF  { v8f v; float f[8]; };
union U4H    { uint4 u; unsigned short h[8]; unsigned int w[4]; };

__device__ __forceinline__ unsigned short f2bf(float f) {
    unsigned int u = __builtin_bit_cast(unsigned int, f);
    u += 0x7FFFu + ((u >> 16) & 1u);           // round-to-nearest-even
    return (unsigned short)(u >> 16);
}
__device__ __forceinline__ float bf2f(unsigned short h) {
    unsigned int u = ((unsigned int)h) << 16;
    return __builtin_bit_cast(float, u);
}

#define WMMA_BF16(A, Bf, C) \
    __builtin_amdgcn_wmma_f32_16x16x32_bf16(false, (A), false, (Bf), (short)0, (C), false, false)

// Load a 16-halfword fragment as two b128s (p0, p1 must be 16B aligned).
__device__ __forceinline__ void ld_frag(FragBF& f, const unsigned short* p0,
                                        const unsigned short* p1) {
    const uint4 x = *(const uint4*)p0;
    const uint4 y = *(const uint4*)p1;
    f.u[0] = x.x; f.u[1] = x.y; f.u[2] = x.z; f.u[3] = x.w;
    f.u[4] = y.x; f.u[5] = y.y; f.u[6] = y.z; f.u[7] = y.w;
}

// ---------------------------------------------------------------------------
// Weight convert fp32 -> bf16, optional QKV row permute (h,d,comp)->(comp,h,d)
// ---------------------------------------------------------------------------
__global__ __launch_bounds__(256) void w2bf_kernel(
    unsigned short* __restrict__ dst, const float* __restrict__ src,
    int rows, int K, int permuteQKV)
{
    const size_t i = (size_t)blockIdx.x * 256 + threadIdx.x;
    if (i >= (size_t)rows * K) return;
    const int j = (int)(i / K), k = (int)(i - (size_t)j * K);
    const int sj = permuteQKV ? ((j & 511) * 3 + (j >> 9)) : j;
    dst[i] = f2bf(src[(size_t)sj * K + k]);
}

// ---------------------------------------------------------------------------
// WMMA GEMM: out = A(bf16,[M,K]) @ Wb(bf16,[Nout,K])^T + bias (+res)
// One wave computes a 16x64 strip (4 N-tiles). All 5 fragments of a K-step
// are loaded before the 4 WMMAs so loads clause together.
// ---------------------------------------------------------------------------
__global__ __launch_bounds__(256) void gemm_bf16_kernel(
    const unsigned short* __restrict__ A, const unsigned short* __restrict__ Wb,
    const float* __restrict__ bias, const float* __restrict__ resid,
    float* __restrict__ outF, unsigned short* __restrict__ outB,
    int M, int Nout, int K)
{
    const int lane   = threadIdx.x & 31;
    const int wave   = threadIdx.x >> 5;
    const int grpsN  = Nout >> 6;                 // 64-wide column groups
    const int tilesM = (M + 15) >> 4;
    const int grp    = blockIdx.x * 8 + wave;
    if (grp >= tilesM * grpsN) return;
    const int tm = grp / grpsN, gn = grp - tm * grpsN;
    const int m0 = tm << 4, n0 = gn << 6;
    const int col16 = lane & 15;
    const int hi    = lane >> 4;
    const int base8 = hi * 8;                     // C row offset
    const int kb16  = hi * 16;                    // B K-slot offset

    int mA = m0 + col16; if (mA >= M) mA = M - 1;
    const unsigned short* arow = A + (size_t)mA * K;
    const unsigned short* wr0 = Wb + (size_t)(n0 + 0  + col16) * K + kb16;
    const unsigned short* wr1 = Wb + (size_t)(n0 + 16 + col16) * K + kb16;
    const unsigned short* wr2 = Wb + (size_t)(n0 + 32 + col16) * K + kb16;
    const unsigned short* wr3 = Wb + (size_t)(n0 + 48 + col16) * K + kb16;

    FragF acc[4];
#pragma unroll
    for (int nt = 0; nt < 4; ++nt)
#pragma unroll
        for (int r = 0; r < 8; ++r) acc[nt].f[r] = 0.f;

#pragma unroll 2
    for (int kk = 0; kk < K; kk += 32) {
        if (kk + 64 < K) __builtin_prefetch(arow + kk + 64, 0, 1);
        FragBF a, b0, b1, b2, b3;
        ld_frag(a,  arow + kk + kb16, arow + kk + kb16 + 8);
        ld_frag(b0, wr0 + kk, wr0 + kk + 8);
        ld_frag(b1, wr1 + kk, wr1 + kk + 8);
        ld_frag(b2, wr2 + kk, wr2 + kk + 8);
        ld_frag(b3, wr3 + kk, wr3 + kk + 8);
        acc[0].v = WMMA_BF16(a.v, b0.v, acc[0].v);
        acc[1].v = WMMA_BF16(a.v, b1.v, acc[1].v);
        acc[2].v = WMMA_BF16(a.v, b2.v, acc[2].v);
        acc[3].v = WMMA_BF16(a.v, b3.v, acc[3].v);
    }

#pragma unroll
    for (int nt = 0; nt < 4; ++nt) {
        const int col = n0 + nt * 16 + col16;
        const float bi = bias ? bias[col] : 0.f;
#pragma unroll
        for (int r = 0; r < 8; ++r) {
            const int row = m0 + r + base8;
            if (row >= M) continue;
            float v = acc[nt].f[r] + bi;
            const size_t idx = (size_t)row * Nout + col;
            if (resid) v += resid[idx];
            if (outF) outF[idx] = v;
            if (outB) outB[idx] = f2bf(v);
        }
    }
}

// ---------------------------------------------------------------------------
// LayerNorm (rows of 256) -> bf16. One wave per row, 8 rows per block.
// ---------------------------------------------------------------------------
__global__ __launch_bounds__(256) void ln_bf16_kernel(
    unsigned short* __restrict__ out, const float* __restrict__ x,
    const float* __restrict__ g, const float* __restrict__ b, int Mrows)
{
    const int lane = threadIdx.x & 31;
    const int row  = blockIdx.x * 8 + (threadIdx.x >> 5);
    if (row >= Mrows) return;
    const float* xr = x + (size_t)row * DIMC;
    float vals[8], s = 0.f, s2 = 0.f;
    const float4 v0 = *(const float4*)(xr + lane * 8);
    const float4 v1 = *(const float4*)(xr + lane * 8 + 4);
    vals[0] = v0.x; vals[1] = v0.y; vals[2] = v0.z; vals[3] = v0.w;
    vals[4] = v1.x; vals[5] = v1.y; vals[6] = v1.z; vals[7] = v1.w;
#pragma unroll
    for (int j = 0; j < 8; ++j) { s += vals[j]; s2 += vals[j] * vals[j]; }
#pragma unroll
    for (int m = 1; m < 32; m <<= 1) { s += __shfl_xor(s, m, 32); s2 += __shfl_xor(s2, m, 32); }
    const float mean = s * (1.f / DIMC);
    const float var  = s2 * (1.f / DIMC) - mean * mean;
    const float inv  = rsqrtf(var + 1e-5f);
    U4H o;
#pragma unroll
    for (int j = 0; j < 8; ++j) {
        const int c = lane * 8 + j;
        o.h[j] = f2bf((vals[j] - mean) * inv * g[c] + b[c]);
    }
    *(uint4*)(out + (size_t)row * DIMC + lane * 8) = o.u;
}

// ---------------------------------------------------------------------------
// Circular roll along n (vectorized float4)
// ---------------------------------------------------------------------------
__global__ __launch_bounds__(256) void roll_kernel(
    float* __restrict__ dst, const float* __restrict__ src, int shift)
{
    const size_t i = (size_t)blockIdx.x * 256 + threadIdx.x;   // B*N*DIM/4
    const int c4 = (int)(i & 63) * 4;
    const size_t bn = i >> 6;
    const int n = (int)(bn & (NN - 1));
    const size_t b = bn >> 13;
    const int ns = (n + shift) & (NN - 1);
    *(float4*)(dst + (bn << 8) + c4) =
        *(const float4*)(src + ((((b << 13) + ns)) << 8) + c4);
}

// ---------------------------------------------------------------------------
// Depthwise conv1d (k=3, zero pad) + bias + ReLU : bf16 in, bf16 out
// ---------------------------------------------------------------------------
__global__ __launch_bounds__(256) void dwconv_relu_kernel(
    unsigned short* __restrict__ out, const unsigned short* __restrict__ h,
    const float* __restrict__ Wdw, const float* __restrict__ bdw)
{
    const size_t i = (size_t)blockIdx.x * 256 + threadIdx.x;   // B*N*512/8
    const int c8 = (int)(i & 63) * 8;
    const size_t bn = i >> 6;
    const int n = (int)(bn & (NN - 1));
    U4H um, ul, ur, o;
    um.u = *(const uint4*)(h + (bn << 9) + c8);
    const bool hasl = (n > 0), hasr = (n < NN - 1);
    if (hasl) ul.u = *(const uint4*)(h + ((bn - 1) << 9) + c8);
    if (hasr) ur.u = *(const uint4*)(h + ((bn + 1) << 9) + c8);
#pragma unroll
    for (int j = 0; j < 8; ++j) {
        const int c = c8 + j;
        const float* w = Wdw + c * 3;
        float acc = bdw[c] + bf2f(um.h[j]) * w[1];
        if (hasl) acc += bf2f(ul.h[j]) * w[0];
        if (hasr) acc += bf2f(ur.h[j]) * w[2];
        o.h[j] = f2bf(acc > 0.f ? acc : 0.f);
    }
    *(uint4*)(out + (bn << 9) + c8) = o.u;
}

// ---------------------------------------------------------------------------
// Enformer-style relative positional features -> bf16 [255][480]
// ---------------------------------------------------------------------------
__global__ __launch_bounds__(256) void pos_kernel(unsigned short* __restrict__ pos)
{
    const int i = blockIdx.x * 256 + threadIdx.x;
    if (i >= PREL * NRPF) return;
    const int p = i / NRPF, f = i - p * NRPF;
    const float d  = (float)(p - (WINT - 1));
    const float ad = fabsf(d);
    int idx = f; float mult = 1.f;
    if (f >= 240) { idx = f - 240; mult = (d > 0.f) ? 1.f : ((d < 0.f) ? -1.f : 0.f); }
    float v;
    if (idx < 80) {
        const float hl = exp2f(3.f + (float)idx * (4.f / 79.f));
        v = expf(-0.69314718056f / hl * ad);
    } else if (idx < 160) {
        const int ii = idx - 80;
        const float cw = exp2f((float)(ii + 1)) - 1.f;
        v = (cw > ad) ? 1.f : 0.f;
    } else {
        const int ii = idx - 160;
        const float stddev = (float)WINT / 160.f;
        const float start_mean = (float)WINT / 80.f;
        const float logx = logf(fmaxf(ad, 1e-20f));
        float mymax = 0.f, mine = 0.f;
        for (int t = 0; t < 80; ++t) {
            const float mean = start_mean + ((float)WINT - start_mean) * (float)t / 79.f;
            const float conc = (mean / stddev) * (mean / stddev);
            const float rate = mean / (stddev * stddev);
            const float logp = (conc - 1.f) * logx - rate * ad
                             - (lgammaf(conc) - conc * logf(rate));
            const float pv = expf(logp) + 1e-8f;
            if (pv > mymax) mymax = pv;
            if (t == ii)    mine  = pv;
        }
        v = mine / mymax;
    }
    pos[i] = f2bf(v * mult);
}

// ---------------------------------------------------------------------------
// Fused windowed attention. 1 block = one (b, head, window); 8 waves, each
// wave owns 16 query rows. qkv planes are (comp,h,d)-ordered: q at +0,
// k at +512, v at +1024 within each row of 1536. A-fragments are loaded
// once and reused across all column tiles.
// ---------------------------------------------------------------------------
__global__ __launch_bounds__(256) void attn_kernel(
    const unsigned short* __restrict__ qkv,   // [B*N, 1536] bf16, (comp,h,d)
    const unsigned short* __restrict__ relk,  // [255, 512] bf16 (this layer)
    const float* __restrict__ rcb,            // [8*64]
    const float* __restrict__ rpb,            // [8*64]
    unsigned short* __restrict__ out)         // [B*N, 512] bf16
{
    __shared__ alignas(16) unsigned short sQc[WINT][DKEY];   // 16 KB
    __shared__ alignas(16) unsigned short sQr[WINT][DKEY];   // 16 KB
    __shared__ alignas(16) unsigned short sK [WINT][DKEY];   // 16 KB
    __shared__ alignas(16) unsigned short sVt[DKEY][WINT];   // 16 KB (transposed)
    __shared__ alignas(16) unsigned short sRel[8][16][256];  // 64 KB

    const int blk = blockIdx.x;                 // ((b*8 + h)*64 + w)
    const int w = blk & 63;
    const int h = (blk >> 6) & 7;
    const int b = blk >> 9;
    const size_t rowbase = ((size_t)b * NN + (size_t)w * WINT);

    // ---- stage: 8 contiguous d per thread, b128 loads from q/k/v planes ----
    for (int e = threadIdx.x; e < WINT * DKEY / 8; e += 256) {
        const int t = e >> 3, d8 = (e & 7) * 8;
        const unsigned short* base = qkv + (rowbase + t) * 1536 + h * DKEY + d8;
        U4H q4, k4, v4, qc, qr;
        q4.u = *(const uint4*)(base);
        k4.u = *(const uint4*)(base + 512);
        v4.u = *(const uint4*)(base + 1024);
        *(uint4*)&sK[t][d8] = k4.u;
#pragma unroll
        for (int j = 0; j < 8; ++j) {
            const float qf = bf2f(q4.h[j]);
            qc.h[j] = f2bf(qf + rcb[h * DKEY + d8 + j]);
            qr.h[j] = f2bf(qf + rpb[h * DKEY + d8 + j]);
            sVt[d8 + j][t] = v4.h[j];
        }
        *(uint4*)&sQc[t][d8] = qc.u;
        *(uint4*)&sQr[t][d8] = qr.u;
    }
    __syncthreads();

    const int lane  = threadIdx.x & 31;
    const int wv    = threadIdx.x >> 5;
    const int col16 = lane & 15;
    const int hi    = lane >> 4;
    const int base8 = hi * 8;
    const int kb16  = hi * 16;
    const int mrow  = wv * 16 + col16;

    // ---- hoisted A-fragments (depend only on K-step) ----
    FragBF aQr0, aQr1, aQc0, aQc1;
    ld_frag(aQr0, &sQr[mrow][kb16],      &sQr[mrow][kb16 + 8]);
    ld_frag(aQr1, &sQr[mrow][32 + kb16], &sQr[mrow][32 + kb16 + 8]);
    ld_frag(aQc0, &sQc[mrow][kb16],      &sQc[mrow][kb16 + 8]);
    ld_frag(aQc1, &sQc[mrow][32 + kb16], &sQc[mrow][32 + kb16 + 8]);

    // ---- relative logits: (q+rpb) @ relk^T -> sRel ----
    for (int pt = 0; pt < 16; ++pt) {
        FragF acc;
#pragma unroll
        for (int r = 0; r < 8; ++r) acc.f[r] = 0.f;
        int p = pt * 16 + col16; if (p > PREL - 1) p = PREL - 1;
        const unsigned short* rk = relk + (size_t)p * HDC + h * DKEY + kb16;
        FragBF bf0, bf1;
        ld_frag(bf0, rk,      rk + 8);
        ld_frag(bf1, rk + 32, rk + 40);
        acc.v = WMMA_BF16(aQr0.v, bf0.v, acc.v);
        acc.v = WMMA_BF16(aQr1.v, bf1.v, acc.v);
#pragma unroll
        for (int r = 0; r < 8; ++r)
            sRel[wv][r + base8][pt * 16 + col16] = f2bf(acc.f[r]);
    }

    // ---- content logits: (q+rcb) @ k^T ----
    FragF acc[8];
#pragma unroll
    for (int nt = 0; nt < 8; ++nt)
#pragma unroll
        for (int r = 0; r < 8; ++r) acc[nt].f[r] = 0.f;

#pragma unroll
    for (int nt = 0; nt < 8; ++nt) {
        const int s = nt * 16 + col16;
        FragBF bf0, bf1;
        ld_frag(bf0, &sK[s][kb16],      &sK[s][kb16 + 8]);
        ld_frag(bf1, &sK[s][32 + kb16], &sK[s][32 + kb16 + 8]);
        acc[nt].v = WMMA_BF16(aQc0.v, bf0.v, acc[nt].v);
        acc[nt].v = WMMA_BF16(aQc1.v, bf1.v, acc[nt].v);
    }
    __syncthreads();

    // ---- rel-shift add, row softmax, probs -> LDS bf16 ----
#pragma unroll
    for (int r = 0; r < 8; ++r) {
        const int rloc = r + base8;
        const int tloc = wv * 16 + rloc;
        float mx = -1e30f;
#pragma unroll
        for (int nt = 0; nt < 8; ++nt) {
            const int s = nt * 16 + col16;
            const float sc = acc[nt].f[r] * 0.125f
                           + bf2f(sRel[wv][rloc][s - tloc + (WINT - 1)]);
            acc[nt].f[r] = sc;
            mx = fmaxf(mx, sc);
        }
#pragma unroll
        for (int m = 1; m < 16; m <<= 1) mx = fmaxf(mx, __shfl_xor(mx, m, 32));
        float sum = 0.f;
#pragma unroll
        for (int nt = 0; nt < 8; ++nt) {
            const float e = expf(acc[nt].f[r] - mx);
            acc[nt].f[r] = e;
            sum += e;
        }
#pragma unroll
        for (int m = 1; m < 16; m <<= 1) sum += __shfl_xor(sum, m, 32);
        const float inv = 1.f / sum;
#pragma unroll
        for (int nt = 0; nt < 8; ++nt)
            sRel[wv][rloc][nt * 16 + col16] = f2bf(acc[nt].f[r] * inv);
    }
    __syncthreads();

    // ---- out = P @ V : prob A-fragments hoisted, V transposed in LDS ----
    FragBF aP0, aP1, aP2, aP3;
    ld_frag(aP0, &sRel[wv][col16][kb16],      &sRel[wv][col16][kb16 + 8]);
    ld_frag(aP1, &sRel[wv][col16][32 + kb16], &sRel[wv][col16][32 + kb16 + 8]);
    ld_frag(aP2, &sRel[wv][col16][64 + kb16], &sRel[wv][col16][64 + kb16 + 8]);
    ld_frag(aP3, &sRel[wv][col16][96 + kb16], &sRel[wv][col16][96 + kb16 + 8]);

#pragma unroll
    for (int nt2 = 0; nt2 < 4; ++nt2) {
        FragF o;
#pragma unroll
        for (int r = 0; r < 8; ++r) o.f[r] = 0.f;
        const int d = nt2 * 16 + col16;
        FragBF b0, b1, b2, b3;
        ld_frag(b0, &sVt[d][kb16],       &sVt[d][kb16 + 8]);
        ld_frag(b1, &sVt[d][32 + kb16],  &sVt[d][32 + kb16 + 8]);
        ld_frag(b2, &sVt[d][64 + kb16],  &sVt[d][64 + kb16 + 8]);
        ld_frag(b3, &sVt[d][96 + kb16],  &sVt[d][96 + kb16 + 8]);
        o.v = WMMA_BF16(aP0.v, b0.v, o.v);
        o.v = WMMA_BF16(aP1.v, b1.v, o.v);
        o.v = WMMA_BF16(aP2.v, b2.v, o.v);
        o.v = WMMA_BF16(aP3.v, b3.v, o.v);
#pragma unroll
        for (int r = 0; r < 8; ++r) {
            const int tloc = wv * 16 + r + base8;
            out[(rowbase + tloc) * HDC + h * DKEY + d] = f2bf(o.f[r]);
        }
    }
}

// ---------------------------------------------------------------------------
// Host orchestration
// ---------------------------------------------------------------------------
extern "C" void kernel_launch(void* const* d_in, const int* in_sizes, int n_in,
                              void* d_out, int out_size, void* d_ws, size_t ws_size,
                              hipStream_t stream)
{
    (void)in_sizes; (void)n_in; (void)out_size; (void)ws_size;
    const float* x_in  = (const float*)d_in[0];
    const float* Wqkv  = (const float*)d_in[1];
    const float* Wout  = (const float*)d_in[2];
    const float* bout  = (const float*)d_in[3];
    const float* Wrel  = (const float*)d_in[4];
    const float* rcb   = (const float*)d_in[5];
    const float* rpb   = (const float*)d_in[6];
    const float* ln1g  = (const float*)d_in[7];
    const float* ln1b  = (const float*)d_in[8];
    const float* ln2g  = (const float*)d_in[9];
    const float* ln2b  = (const float*)d_in[10];
    const float* W1    = (const float*)d_in[11];
    const float* b1    = (const float*)d_in[12];
    const float* Wdw   = (const float*)d_in[13];
    const float* bdw   = (const float*)d_in[14];
    const float* W2    = (const float*)d_in[15];
    const float* b2    = (const float*)d_in[16];

    const size_t Mrows = (size_t)BB * NN;            // 65536
    size_t off = 0;
    auto alloc = [&](size_t bytes) -> void* {
        void* p = (char*)d_ws + off;
        off += (bytes + 255) & ~(size_t)255;
        return p;
    };
    float*          x_cur = (float*)alloc(Mrows * DIMC * 4);
    float*          x_alt = (float*)alloc(Mrows * DIMC * 4);
    unsigned short* xln   = (unsigned short*)alloc(Mrows * DIMC * 2);
    unsigned short* qkv   = (unsigned short*)alloc(Mrows * 1536 * 2);
    unsigned short* attnb = (unsigned short*)alloc(Mrows * HDC * 2);
    unsigned short* hbuf  = (unsigned short*)alloc(Mrows * HDC * 2);
    unsigned short* hc    = (unsigned short*)alloc(Mrows * HDC * 2);
    unsigned short* posb  = (unsigned short*)alloc((size_t)PREL * NRPF * 2);
    unsigned short* relk  = (unsigned short*)alloc((size_t)NLAY * PREL * HDC * 2);
    unsigned short* bWqkv = (unsigned short*)alloc((size_t)NLAY * 1536 * DIMC * 2);
    unsigned short* bWout = (unsigned short*)alloc((size_t)NLAY * DIMC * HDC * 2);
    unsigned short* bW1   = (unsigned short*)alloc((size_t)NLAY * HDC * DIMC * 2);
    unsigned short* bW2   = (unsigned short*)alloc((size_t)NLAY * DIMC * HDC * 2);
    unsigned short* bWrel = (unsigned short*)alloc((size_t)NLAY * HDC * NRPF * 2);

    // ---- weight conversion (once per call) ----
    for (int l = 0; l < NLAY; ++l) {
        w2bf_kernel<<<(1536 * DIMC + 255) / 256, 256, 0, stream>>>(
            bWqkv + (size_t)l * 1536 * DIMC, Wqkv + (size_t)l * 1536 * DIMC, 1536, DIMC, 1);
        w2bf_kernel<<<(DIMC * HDC + 255) / 256, 256, 0, stream>>>(
            bWout + (size_t)l * DIMC * HDC, Wout + (size_t)l * DIMC * HDC, DIMC, HDC, 0);
        w2bf_kernel<<<(HDC * DIMC + 255) / 256, 256, 0, stream>>>(
            bW1 + (size_t)l * HDC * DIMC, W1 + (size_t)l * HDC * DIMC, HDC, DIMC, 0);
        w2bf_kernel<<<(DIMC * HDC + 255) / 256, 256, 0, stream>>>(
            bW2 + (size_t)l * DIMC * HDC, W2 + (size_t)l * DIMC * HDC, DIMC, HDC, 0);
        w2bf_kernel<<<(HDC * NRPF + 255) / 256, 256, 0, stream>>>(
            bWrel + (size_t)l * HDC * NRPF, Wrel + (size_t)l * HDC * NRPF, HDC, NRPF, 0);
    }

    // ---- positional features + per-layer rel_k ----
    pos_kernel<<<(PREL * NRPF + 255) / 256, 256, 0, stream>>>(posb);
    for (int l = 0; l < NLAY; ++l) {
        const int groups = 16 * (HDC / 64);          // ceil(255/16) * 8 = 128
        gemm_bf16_kernel<<<groups / 8, 256, 0, stream>>>(
            posb, bWrel + (size_t)l * HDC * NRPF, nullptr, nullptr,
            nullptr, relk + (size_t)l * PREL * HDC, PREL, HDC, NRPF);
    }

    hipMemcpyAsync(x_cur, x_in, Mrows * DIMC * 4, hipMemcpyDeviceToDevice, stream);

    for (int l = 0; l < NLAY; ++l) {
        // ---- attention residual ----
        ln_bf16_kernel<<<Mrows / 8, 256, 0, stream>>>(
            xln, x_cur, ln1g + l * DIMC, ln1b + l * DIMC, (int)Mrows);
        {   // QKV: [65536,256] x [1536,256]^T -> bf16 (comp,h,d planes)
            const int groups = (int)(Mrows / 16) * (1536 / 64);
            gemm_bf16_kernel<<<groups / 8, 256, 0, stream>>>(
                xln, bWqkv + (size_t)l * 1536 * DIMC, nullptr, nullptr,
                nullptr, qkv, (int)Mrows, 1536, DIMC);
        }
        attn_kernel<<<BB * HEADS * NWIN, 256, 0, stream>>>(
            qkv, relk + (size_t)l * PREL * HDC,
            rcb + (size_t)l * HEADS * DKEY, rpb + (size_t)l * HEADS * DKEY, attnb);
        {   // out-proj + residual into x_cur
            const int groups = (int)(Mrows / 16) * (DIMC / 64);
            gemm_bf16_kernel<<<groups / 8, 256, 0, stream>>>(
                attnb, bWout + (size_t)l * DIMC * HDC, bout + l * DIMC, x_cur,
                x_cur, nullptr, (int)Mrows, DIMC, HDC);
        }

        // ---- conv-FFN residual ----
        ln_bf16_kernel<<<Mrows / 8, 256, 0, stream>>>(
            xln, x_cur, ln2g + l * DIMC, ln2b + l * DIMC, (int)Mrows);
        {   // W1: 256 -> 512 (bf16 out for conv)
            const int groups = (int)(Mrows / 16) * (HDC / 64);
            gemm_bf16_kernel<<<groups / 8, 256, 0, stream>>>(
                xln, bW1 + (size_t)l * HDC * DIMC, b1 + l * HDC, nullptr,
                nullptr, hbuf, (int)Mrows, HDC, DIMC);
        }
        dwconv_relu_kernel<<<(unsigned)(Mrows * HDC / 8 / 256), 256, 0, stream>>>(
            hc, hbuf, Wdw + (size_t)l * HDC * 3, bdw + l * HDC);
        {   // W2: 512 -> 256 + residual into x_cur
            const int groups = (int)(Mrows / 16) * (DIMC / 64);
            gemm_bf16_kernel<<<groups / 8, 256, 0, stream>>>(
                hc, bW2 + (size_t)l * DIMC * HDC, b2 + l * DIMC, x_cur,
                x_cur, nullptr, (int)Mrows, DIMC, HDC);
        }

        // ---- circular roll between layers ----
        const int shift = (l & 1) ? (NN - HALFW) : HALFW;  // roll(-64) / roll(+64)
        roll_kernel<<<(unsigned)(Mrows * DIMC / 4 / 256), 256, 0, stream>>>(x_alt, x_cur, shift);
        float* t = x_cur; x_cur = x_alt; x_alt = t;
    }

    hipMemcpyAsync(d_out, x_cur, Mrows * DIMC * 4, hipMemcpyDeviceToDevice, stream);
}
